// PointNet2BackboneLight_8091718385924
// MI455X (gfx1250) — matrix operand order repaired and checked by reading, not compiled, not run on hardware
//
#include <hip/hip_runtime.h>
#include <hip/hip_bf16.h>

// ---------------------------------------------------------------------------
// PointNet++ backbone for MI455X (gfx1250, wave32, WMMA).
// All MLP / 1x1-conv layers run a v_wmma_f32_16x16x32_f16 GEMM with f16
// pre-padded staging (branch-free b128 fragment loads), f32 accumulate,
// fused bias+BN+ReLU epilogue. Requires rows%32==0, Cout%32==0, Kpad%32==0
// (true for every layer in this network).
// ---------------------------------------------------------------------------

typedef _Float16 h16;
typedef __attribute__((ext_vector_type(8)))  _Float16 v8h;
typedef __attribute__((ext_vector_type(16))) _Float16 v16h;
typedef __attribute__((ext_vector_type(8)))  float    v8f;

#define BN_INV 0.99999500003749968f  // 1/sqrt(1 + 1e-5), eval-mode BatchNorm

// ---------------------------------------------------------------------------
// Furthest point sampling: one block per batch, dist[] in LDS, argmax via
// LDS tree reduction (ties -> lower index, matching jnp.argmax).
// ---------------------------------------------------------------------------
__global__ void fps_kernel(const float* __restrict__ xyz, int* __restrict__ fidx,
                           int N, int npoint) {
  __shared__ float s_dist[4096];
  __shared__ float s_rv[256];
  __shared__ int   s_ri[256];
  __shared__ int   s_last;
  const int b = blockIdx.x;
  const int t = threadIdx.x;
  const float* px = xyz + (size_t)b * N * 3;
  for (int i = t; i < N; i += 256) s_dist[i] = 1e10f;
  if (t == 0) { fidx[b * npoint] = 0; s_last = 0; }
  __syncthreads();
  for (int it = 1; it < npoint; ++it) {
    const int last = s_last;
    const float lx = px[last * 3 + 0];
    const float ly = px[last * 3 + 1];
    const float lz = px[last * 3 + 2];
    float bv = -1.0f; int bi = 0;
    for (int i = t; i < N; i += 256) {
      float dx = px[i * 3 + 0] - lx;
      float dy = px[i * 3 + 1] - ly;
      float dz = px[i * 3 + 2] - lz;
      float d  = dx * dx + dy * dy + dz * dz;
      float nd = fminf(s_dist[i], d);
      s_dist[i] = nd;
      if (nd > bv) { bv = nd; bi = i; }
    }
    s_rv[t] = bv; s_ri[t] = bi;
    __syncthreads();
    for (int off = 128; off > 0; off >>= 1) {
      if (t < off) {
        float ov = s_rv[t + off]; int oi = s_ri[t + off];
        if (ov > s_rv[t] || (ov == s_rv[t] && oi < s_ri[t])) {
          s_rv[t] = ov; s_ri[t] = oi;
        }
      }
      __syncthreads();
    }
    if (t == 0) { s_last = s_ri[0]; fidx[b * npoint + it] = s_ri[0]; }
    __syncthreads();
  }
}

__global__ void gather_xyz_kernel(const float* __restrict__ xyz,
                                  const int* __restrict__ fidx,
                                  float* __restrict__ nxyz, int N, int S, int B) {
  int tid = blockIdx.x * blockDim.x + threadIdx.x;
  if (tid >= B * S) return;
  int b  = tid / S;
  int fi = fidx[tid];
  const float* p = xyz + ((size_t)b * N + fi) * 3;
  nxyz[tid * 3 + 0] = p[0];
  nxyz[tid * 3 + 1] = p[1];
  nxyz[tid * 3 + 2] = p[2];
}

// First 16 points (index order) within radius; pad with first hit
// (or N-1 when no hits), matching the sorted-key reference semantics.
__global__ void ball_query_kernel(const float* __restrict__ xyz,
                                  const float* __restrict__ nxyz,
                                  int* __restrict__ gidx,
                                  int N, int S, int B, float r2) {
  int tid = blockIdx.x * blockDim.x + threadIdx.x;
  if (tid >= B * S) return;
  int b = tid / S;
  const float* px = xyz + (size_t)b * N * 3;
  float qx = nxyz[tid * 3 + 0];
  float qy = nxyz[tid * 3 + 1];
  float qz = nxyz[tid * 3 + 2];
  int hits[16];
  int cnt = 0;
  for (int i = 0; i < N && cnt < 16; ++i) {
    float dx = px[i * 3 + 0] - qx;
    float dy = px[i * 3 + 1] - qy;
    float dz = px[i * 3 + 2] - qz;
    if (dx * dx + dy * dy + dz * dz < r2) hits[cnt++] = i;
  }
  int pad = (cnt > 0) ? hits[0] : (N - 1);
  for (int k = cnt; k < 16; ++k) hits[k] = pad;
  for (int k = 0; k < 16; ++k) gidx[tid * 16 + k] = hits[k];
}

// Grouped SA input rows (f16, zero-padded to Kpad): [rel_xyz(3)|feats(Cf)|0...]
__global__ void build_group_kernel(const float* __restrict__ xyz,
                                   const float* __restrict__ nxyz,
                                   const h16* __restrict__ feats,
                                   const int* __restrict__ gidx,
                                   h16* __restrict__ out,
                                   int N, int S, int Cf, int Kpad, int B) {
  int tid = blockIdx.x * blockDim.x + threadIdx.x;
  int total = B * S * 16;
  if (tid >= total) return;
  int bs  = tid >> 4;
  int b   = bs / S;
  int src = gidx[tid];
  h16* o = out + (size_t)tid * Kpad;
  const float* p = xyz  + ((size_t)b * N + src) * 3;
  const float* q = nxyz + (size_t)bs * 3;
  o[0] = (h16)(p[0] - q[0]);
  o[1] = (h16)(p[1] - q[1]);
  o[2] = (h16)(p[2] - q[2]);
  const h16* f = feats + ((size_t)b * N + src) * Cf;
  for (int c = 0; c < Cf; ++c) o[3 + c] = f[c];
  for (int c = 3 + Cf; c < Kpad; ++c) o[c] = (h16)0.0f;
}

// Max over the 16 group samples -> point-major [B*S, C] (f16).
__global__ void maxpool_kernel(const h16* __restrict__ pre,
                               h16* __restrict__ out, int BS, int C) {
  int tid = blockIdx.x * blockDim.x + threadIdx.x;
  if (tid >= BS * C) return;
  int o  = tid % C;
  int bs = tid / C;
  const h16* p = pre + ((size_t)bs * 16) * C + o;
  float m = (float)p[0];
  for (int g = 1; g < 16; ++g) m = fmaxf(m, (float)p[(size_t)g * C]);
  out[(size_t)bs * C + o] = (h16)m;
}

// 3-NN (squared distances) + normalized inverse-distance weights.
__global__ void three_nn_kernel(const float* __restrict__ uxyz,
                                const float* __restrict__ kxyz,
                                int* __restrict__ idx3, float* __restrict__ w3,
                                int Nu, int Nk, int B) {
  int tid = blockIdx.x * blockDim.x + threadIdx.x;
  if (tid >= B * Nu) return;
  int b = tid / Nu;
  float qx = uxyz[tid * 3 + 0];
  float qy = uxyz[tid * 3 + 1];
  float qz = uxyz[tid * 3 + 2];
  const float* pk = kxyz + (size_t)b * Nk * 3;
  float d0 = 1e30f, d1 = 1e30f, d2v = 1e30f;
  int   i0 = 0,     i1 = 0,     i2  = 0;
  for (int i = 0; i < Nk; ++i) {
    float dx = pk[i * 3 + 0] - qx;
    float dy = pk[i * 3 + 1] - qy;
    float dz = pk[i * 3 + 2] - qz;
    float d  = dx * dx + dy * dy + dz * dz;
    if (d < d0)       { d2v = d1; i2 = i1; d1 = d0; i1 = i0; d0 = d; i0 = i; }
    else if (d < d1)  { d2v = d1; i2 = i1; d1 = d;  i1 = i; }
    else if (d < d2v) { d2v = d;  i2 = i; }
  }
  float w0 = 1.0f / (d0 + 1e-8f);
  float w1 = 1.0f / (d1 + 1e-8f);
  float w2 = 1.0f / (d2v + 1e-8f);
  float ws = w0 + w1 + w2;
  w3[tid * 3 + 0] = w0 / ws;
  w3[tid * 3 + 1] = w1 / ws;
  w3[tid * 3 + 2] = w2 / ws;
  idx3[tid * 3 + 0] = i0;
  idx3[tid * 3 + 1] = i1;
  idx3[tid * 3 + 2] = i2;
}

// FP input rows (f16, zero-padded): [interp known (Ck) | unknown (Cu) | 0...]
__global__ void build_fp_kernel(const h16* __restrict__ knf,
                                const h16* __restrict__ unkf,
                                const int* __restrict__ idx3,
                                const float* __restrict__ w3,
                                h16* __restrict__ out,
                                int Nu, int Nk, int Ck, int Cu, int Kpad, int B) {
  int tid = blockIdx.x * blockDim.x + threadIdx.x;
  if (tid >= B * Nu) return;
  int b = tid / Nu;
  int i0 = idx3[tid * 3 + 0], i1 = idx3[tid * 3 + 1], i2 = idx3[tid * 3 + 2];
  float w0 = w3[tid * 3 + 0], w1 = w3[tid * 3 + 1], w2 = w3[tid * 3 + 2];
  const h16* f0 = knf + ((size_t)b * Nk + i0) * Ck;
  const h16* f1 = knf + ((size_t)b * Nk + i1) * Ck;
  const h16* f2 = knf + ((size_t)b * Nk + i2) * Ck;
  h16* o = out + (size_t)tid * Kpad;
  for (int c = 0; c < Ck; ++c)
    o[c] = (h16)(w0 * (float)f0[c] + w1 * (float)f1[c] + w2 * (float)f2[c]);
  const h16* u = unkf + (size_t)tid * Cu;
  for (int c = 0; c < Cu; ++c) o[Ck + c] = u[c];
  for (int c = Ck + Cu; c < Kpad; ++c) o[c] = (h16)0.0f;
}

// f32 -> f16 elementwise (for input feats).
__global__ void f32_to_f16_kernel(const float* __restrict__ in,
                                  h16* __restrict__ out, int n) {
  int tid = blockIdx.x * blockDim.x + threadIdx.x;
  if (tid < n) out[tid] = (h16)in[tid];
}

// Weight [Cout, Cin] f32 -> [Cout, Kpad] f16, zero-padded.
__global__ void weight_to_f16_kernel(const float* __restrict__ w,
                                     h16* __restrict__ wh,
                                     int Cout, int Cin, int Kpad) {
  int tid = blockIdx.x * blockDim.x + threadIdx.x;
  if (tid >= Cout * Kpad) return;
  int o = tid / Kpad;
  int k = tid - o * Kpad;
  wh[tid] = (k < Cin) ? (h16)w[(size_t)o * Cin + k] : (h16)0.0f;
}

// ---------------------------------------------------------------------------
// WMMA GEMM, 32x32 output super-tile per wave (2x2 of 16x16), 4 wmma/K-chunk.
// Branch-free vector fragment loads per CDNA5 ISA 7.12.2 layouts:
//   A lane(l): M=l&15, K pairs at kb+(l>=16?8:0)+{0..7} and +16  -> 2x b128
//   B lane(l): N=l&15, K pairs at kb+(l>=16?8:0)+{0..7} and +16  -> 2x b128
//   C/D lane(l): M = vgpr_j + (l>=16?8:0), N = l&15
// OUT = relu((acc + bias[o]) * gamma[o]*BN_INV + beta[o]); f32 or f16 dest.
// ---------------------------------------------------------------------------
__global__ void wmma_mlp_kernel(const h16* __restrict__ IN,
                                const h16* __restrict__ W,
                                const float* __restrict__ gamma,
                                const float* __restrict__ beta,
                                const float* __restrict__ bias,
                                float* __restrict__ outF,
                                h16* __restrict__ outH,
                                int rows, int Kpad, int Cout) {
  const int lane = threadIdx.x & 31;
  const int wave = threadIdx.x >> 5;
  const int nt2  = Cout >> 5;
  const int tile = blockIdx.x * 4 + wave;              // wave-uniform
  const int mt   = tile / nt2;
  const int nt   = tile - mt * nt2;
  if (mt >= (rows >> 5)) return;                       // whole wave exits
  const int row0 = mt << 5;
  const int col0 = nt << 5;

  const int mA    = lane & 15;
  const int koffA = (lane >> 4) << 3;                  // 0 or 8
  const int nB    = lane & 15;
  const int koffB = (lane >> 4) << 3;                  // 0 or 8

  const h16* aRow0 = IN + (size_t)(row0 + mA) * Kpad + koffA;
  const h16* aRow1 = aRow0 + (size_t)16 * Kpad;
  const h16* bCol0 = W  + (size_t)(col0 + nB) * Kpad + koffB;
  const h16* bCol1 = bCol0 + (size_t)16 * Kpad;

  v8f acc00 = {}, acc01 = {}, acc10 = {}, acc11 = {};
  for (int kb = 0; kb < Kpad; kb += 32) {
    v8h a0l = *(const v8h*)(aRow0 + kb);
    v8h a0h = *(const v8h*)(aRow0 + kb + 16);
    v8h a1l = *(const v8h*)(aRow1 + kb);
    v8h a1h = *(const v8h*)(aRow1 + kb + 16);
    v16h b0 = __builtin_shufflevector(*(const v8h*)(bCol0 + kb), *(const v8h*)(bCol0 + kb + 16),
                                      0, 1, 2, 3, 4, 5, 6, 7, 8, 9, 10, 11, 12, 13, 14, 15);
    v16h b1 = __builtin_shufflevector(*(const v8h*)(bCol1 + kb), *(const v8h*)(bCol1 + kb + 16),
                                      0, 1, 2, 3, 4, 5, 6, 7, 8, 9, 10, 11, 12, 13, 14, 15);
    v16h a0 = __builtin_shufflevector(a0l, a0h, 0, 1, 2, 3, 4, 5, 6, 7,
                                      8, 9, 10, 11, 12, 13, 14, 15);
    v16h a1 = __builtin_shufflevector(a1l, a1h, 0, 1, 2, 3, 4, 5, 6, 7,
                                      8, 9, 10, 11, 12, 13, 14, 15);
    acc00 = __builtin_amdgcn_wmma_f32_16x16x32_f16(false, a0, false, b0,
                                                   (short)0, acc00, false, false);
    acc01 = __builtin_amdgcn_wmma_f32_16x16x32_f16(false, a0, false, b1,
                                                   (short)0, acc01, false, false);
    acc10 = __builtin_amdgcn_wmma_f32_16x16x32_f16(false, a1, false, b0,
                                                   (short)0, acc10, false, false);
    acc11 = __builtin_amdgcn_wmma_f32_16x16x32_f16(false, a1, false, b1,
                                                   (short)0, acc11, false, false);
  }

  const int moff = (lane >> 4) << 3;
  const int o0 = col0 + nB;
  const int o1 = o0 + 16;
  const float sc0 = gamma[o0] * BN_INV, sh0 = beta[o0];
  const float sc1 = gamma[o1] * BN_INV, sh1 = beta[o1];
  const float bs0 = bias ? bias[o0] : 0.0f;
  const float bs1 = bias ? bias[o1] : 0.0f;

  if (outF) {
#pragma unroll
    for (int j = 0; j < 8; ++j) {
      const size_t r0 = (size_t)(row0 + moff + j) * Cout;
      const size_t r1 = r0 + (size_t)16 * Cout;
      outF[r0 + o0] = fmaxf((acc00[j] + bs0) * sc0 + sh0, 0.0f);
      outF[r0 + o1] = fmaxf((acc01[j] + bs1) * sc1 + sh1, 0.0f);
      outF[r1 + o0] = fmaxf((acc10[j] + bs0) * sc0 + sh0, 0.0f);
      outF[r1 + o1] = fmaxf((acc11[j] + bs1) * sc1 + sh1, 0.0f);
    }
  } else {
#pragma unroll
    for (int j = 0; j < 8; ++j) {
      const size_t r0 = (size_t)(row0 + moff + j) * Cout;
      const size_t r1 = r0 + (size_t)16 * Cout;
      outH[r0 + o0] = (h16)fmaxf((acc00[j] + bs0) * sc0 + sh0, 0.0f);
      outH[r0 + o1] = (h16)fmaxf((acc01[j] + bs1) * sc1 + sh1, 0.0f);
      outH[r1 + o0] = (h16)fmaxf((acc10[j] + bs0) * sc0 + sh0, 0.0f);
      outH[r1 + o1] = (h16)fmaxf((acc11[j] + bs1) * sc1 + sh1, 0.0f);
    }
  }
}

// ---------------------------------------------------------------------------
// Host-side orchestration (all launches on `stream`).
// ---------------------------------------------------------------------------
static inline int pad32(int x) { return (x + 31) & ~31; }

static void launch_mlp(const h16* in, const h16* wt, const float* g,
                       const float* be, const float* bias,
                       float* outF, h16* outH,
                       int rows, int Kpad, int Cout, hipStream_t s) {
  int tiles  = (rows / 32) * (Cout / 32);
  int blocks = (tiles + 3) / 4;
  wmma_mlp_kernel<<<blocks, 128, 0, s>>>(in, wt, g, be, bias, outF, outH,
                                         rows, Kpad, Cout);
}

static void run_sa(const float* src_xyz, const h16* src_feats, int N, int S, int Cf,
                   float radius,
                   const h16* w0, const float* g0, const float* b0, int C0,
                   const h16* w1, const float* g1, const float* b1, int C1,
                   float* new_xyz, h16* out_feats,
                   int* fidx, int* gidx, h16* gin, h16* mid, h16* pre,
                   hipStream_t s) {
  const int B = 16;
  const int Kp0 = pad32(3 + Cf);
  fps_kernel<<<B, 256, 0, s>>>(src_xyz, fidx, N, S);
  gather_xyz_kernel<<<(B * S + 255) / 256, 256, 0, s>>>(src_xyz, fidx, new_xyz, N, S, B);
  ball_query_kernel<<<(B * S + 255) / 256, 256, 0, s>>>(src_xyz, new_xyz, gidx,
                                                        N, S, B, radius * radius);
  int rows = B * S * 16;
  build_group_kernel<<<(rows + 255) / 256, 256, 0, s>>>(src_xyz, new_xyz, src_feats,
                                                        gidx, gin, N, S, Cf, Kp0, B);
  launch_mlp(gin, w0, g0, b0, nullptr, nullptr, mid, rows, Kp0, C0, s);
  launch_mlp(mid, w1, g1, b1, nullptr, nullptr, pre, rows, C0, C1, s);
  maxpool_kernel<<<(B * S * C1 + 255) / 256, 256, 0, s>>>(pre, out_feats, B * S, C1);
}

static void run_fp(const float* unk_xyz, const float* kn_xyz,
                   const h16* unk_f, int Cu, const h16* kn_f, int Ck,
                   int Nu, int Nk,
                   const h16* w0, const float* g0, const float* b0, int C0,
                   const h16* w1, const float* g1, const float* b1, int C1,
                   h16* out_f, int* idx3, float* w3, h16* fin, h16* mid,
                   hipStream_t s) {
  const int B = 16;
  const int Kp0 = pad32(Ck + Cu);
  three_nn_kernel<<<(B * Nu + 255) / 256, 256, 0, s>>>(unk_xyz, kn_xyz, idx3, w3,
                                                       Nu, Nk, B);
  build_fp_kernel<<<(B * Nu + 255) / 256, 256, 0, s>>>(kn_f, unk_f, idx3, w3, fin,
                                                       Nu, Nk, Ck, Cu, Kp0, B);
  int rows = B * Nu;
  launch_mlp(fin, w0, g0, b0, nullptr, nullptr, mid, rows, Kp0, C0, s);
  launch_mlp(mid, w1, g1, b1, nullptr, nullptr, out_f, rows, C0, C1, s);
}

extern "C" void kernel_launch(void* const* d_in, const int* in_sizes, int n_in,
                              void* d_out, int out_size, void* d_ws, size_t ws_size,
                              hipStream_t stream) {
  (void)in_sizes; (void)n_in; (void)out_size; (void)ws_size;
  const int B = 16, N = 4096;

  const float* xyz = (const float*)d_in[0];
  const float* P[42];
  for (int i = 0; i < 42; ++i) P[i] = (const float*)d_in[i];
  // sa1: 2..7  sa2: 8..13  sa3: 14..19  fp3: 20..25  fp2: 26..31  fp1: 32..37
  // fin_w=38 fin_b=39 fin_g=40 fin_be=41

  // ---- workspace bump allocator ----
  char* base = (char*)d_ws;
  size_t off = 0;
  auto AF = [&](size_t n) -> float* {
    float* p = (float*)(base + off);
    off = (off + n * sizeof(float) + 255) & ~(size_t)255;
    return p;
  };
  auto AI = [&](size_t n) -> int* {
    int* p = (int*)(base + off);
    off = (off + n * sizeof(int) + 255) & ~(size_t)255;
    return p;
  };
  auto AH = [&](size_t n) -> h16* {
    h16* p = (h16*)(base + off);
    off = (off + n * sizeof(h16) + 255) & ~(size_t)255;
    return p;
  };

  float* l1x  = AF((size_t)B * 256 * 3);
  float* l2x  = AF((size_t)B * 64 * 3);
  float* l3x  = AF((size_t)B * 16 * 3);
  h16* featsH = AH((size_t)B * N * 3);
  h16* l1f    = AH((size_t)B * 256 * 128);
  h16* l2f    = AH((size_t)B * 64 * 256);
  h16* l3f    = AH((size_t)B * 16 * 512);
  h16* l2fp   = AH((size_t)B * 64 * 256);
  h16* l1fp   = AH((size_t)B * 256 * 128);
  h16* l0fp   = AH((size_t)B * 4096 * 128);
  int*   fidx = AI((size_t)B * 256);
  int*   gidx = AI((size_t)B * 256 * 16);
  int*   idx3 = AI((size_t)B * 4096 * 3);
  float* w3   = AF((size_t)B * 4096 * 3);
  h16* bufA   = AH((size_t)65536 * 160);   // stage inputs (max: FP1 Kpad=160)
  h16* bufB   = AH((size_t)65536 * 128);   // mid layer outputs
  h16* bufC   = AH((size_t)65536 * 128);   // pre-pool SA outputs

  // ---- weights: f32 [Cout,Cin] -> f16 [Cout,Kpad], zero-padded ----
  struct { int wi, cout, cin; } LW[13] = {
      {2, 64, 6},    {5, 128, 64},   {8, 128, 131},  {11, 256, 128},
      {14, 256, 259},{17, 512, 256}, {20, 256, 768}, {23, 256, 256},
      {26, 128, 384},{29, 128, 128}, {32, 128, 131}, {35, 128, 128},
      {38, 512, 128}};
  h16* WH[13];
  for (int i = 0; i < 13; ++i) {
    int kp = pad32(LW[i].cin);
    WH[i] = AH((size_t)LW[i].cout * kp);
    int n = LW[i].cout * kp;
    weight_to_f16_kernel<<<(n + 255) / 256, 256, 0, stream>>>(
        P[LW[i].wi], WH[i], LW[i].cout, LW[i].cin, kp);
  }
  f32_to_f16_kernel<<<(B * N * 3 + 255) / 256, 256, 0, stream>>>(P[1], featsH,
                                                                 B * N * 3);

  // ---- SA1: N=4096 -> S=256, r=0.04, (6->64),(64->128) ----
  run_sa(xyz, featsH, N, 256, 3, 0.04f,
         WH[0], P[3], P[4], 64, WH[1], P[6], P[7], 128,
         l1x, l1f, fidx, gidx, bufA, bufB, bufC, stream);

  // ---- SA2: 256 -> 64, r=0.08, (131->128),(128->256) ----
  run_sa(l1x, l1f, 256, 64, 128, 0.08f,
         WH[2], P[9], P[10], 128, WH[3], P[12], P[13], 256,
         l2x, l2f, fidx, gidx, bufA, bufB, bufC, stream);

  // ---- SA3: 64 -> 16, r=0.16, (259->256),(256->512) ----
  run_sa(l2x, l2f, 64, 16, 256, 0.16f,
         WH[4], P[15], P[16], 256, WH[5], P[18], P[19], 512,
         l3x, l3f, fidx, gidx, bufA, bufB, bufC, stream);

  // ---- FP3: l3 (16 pts, 512ch) -> l2 (64 pts), (768->256),(256->256) ----
  run_fp(l2x, l3x, l2f, 256, l3f, 512, 64, 16,
         WH[6], P[21], P[22], 256, WH[7], P[24], P[25], 256,
         l2fp, idx3, w3, bufA, bufB, stream);

  // ---- FP2: l2' (64 pts, 256ch) -> l1 (256 pts), (384->128),(128->128) ----
  run_fp(l1x, l2x, l1f, 128, l2fp, 256, 256, 64,
         WH[8], P[27], P[28], 128, WH[9], P[30], P[31], 128,
         l1fp, idx3, w3, bufA, bufB, stream);

  // ---- FP1: l1' (256 pts, 128ch) -> l0 (4096 pts), (131->128),(128->128) ----
  run_fp(xyz, l1x, featsH, 3, l1fp, 128, 4096, 256,
         WH[10], P[33], P[34], 128, WH[11], P[36], P[37], 128,
         l0fp, idx3, w3, bufA, bufB, stream);

  // ---- final: Conv1d(128->512, bias) + BN + ReLU -> d_out [B*N, 512] f32 ----
  launch_mlp(l0fp, WH[12], P[40], P[41], P[39], (float*)d_out, nullptr,
             B * N, 128, 512, stream);
}
